// ProteinFeatures_29317446762976
// MI455X (gfx1250) — matrix operand order repaired and checked by reading, not compiled, not run on hardware
//
#include <hip/hip_runtime.h>
#include <hip/hip_bf16.h>

// ---------------------------------------------------------------------------
// Types
// ---------------------------------------------------------------------------
typedef __bf16 bf16_t;
typedef __attribute__((ext_vector_type(16))) __bf16 v16bf;
typedef __attribute__((ext_vector_type(8)))  __bf16 v8bf;
typedef __attribute__((ext_vector_type(8)))  float  v8f;

// ---------------------------------------------------------------------------
// Problem constants
// ---------------------------------------------------------------------------
#define LRES   1024
#define KNB    30
#define ROWS_E (LRES * KNB)        // 30720
#define KE     416                 // 16 pos + 25*16 rbf (multiple of 32)
#define CTX    25
#define ROWS_V (LRES * CTX)        // 25600
#define KV     192                 // 185 padded to 192
#define ROWS_JE (LRES * CTX * CTX) // 640000
#define KJ     32                  // 16 padded to 32
#define NF     128

// output offsets (in floats), tuple order: E, E_idx, V, J_nodes_f, J_edges_f
#define OFF_E     0LL
#define OFF_EIDX  (OFF_E    + (long long)ROWS_E * NF)     // 3,932,160
#define OFF_V     (OFF_EIDX + (long long)ROWS_E)          // 3,962,880
#define OFF_JN    (OFF_V    + (long long)ROWS_V * NF)     // 7,239,680
#define OFF_JE    (OFF_JN   + (long long)ROWS_V * NF)     // 10,516,480

__constant__ int c_dnarna[26] = {3,2,2,2,0,0,2,0,2,0,2,0,1,0,1,1,0,0,0,1,0,1,2,2,1,0};
// pair table (atom codes 0=N 1=Ca 2=C 3=O 4=Cb); entry 0 = Ca-Ca (uses D_neighbors)
__constant__ int c_pa[25] = {1,0,2,3,4,1,1,1,1,0,0,0,4,4,3,0,2,3,4,2,3,4,2,3,2};
__constant__ int c_pb[25] = {1,0,2,3,4,0,2,3,4,2,3,4,2,3,2,1,1,1,1,0,0,0,4,4,3};

// ---------------------------------------------------------------------------
// Helpers
// ---------------------------------------------------------------------------
union AF { v16bf v; v8bf h[2]; bf16_t e[16]; };

__device__ __forceinline__ float3 get_atom(const float* __restrict__ X,
                                           const float* __restrict__ Cb,
                                           int res, int code) {
  float3 r;
  if (code == 4) {
    const float* p = Cb + res * 3;
    r.x = p[0]; r.y = p[1]; r.z = p[2];
  } else {
    int slot = (code == 3) ? 4 : code;  // N,Ca,C at X[:,0..2], O at X[:,4]
    const float* p = X + (long long)res * 15 + slot * 3;
    r.x = p[0]; r.y = p[1]; r.z = p[2];
  }
  return r;
}

__device__ __forceinline__ float rbf_val(float d, int m) {
  float t = (d - (2.0f + 1.25f * (float)m)) * 0.8f;  // sigma = 1.25
  return __expf(-t * t);
}

// Fused LayerNorm epilogue on a 16x128 wave strip held in 8 v8f accumulators.
// C/D layout: VGPR r, lanes 0-15 -> M=r, lanes 16-31 -> M=r+8; N = 16*t + (lane&15).
__device__ __forceinline__ void ln_store_strip(float* __restrict__ outBase, int rowBase,
                                               const v8f acc[8],
                                               const float* __restrict__ g,
                                               const float* __restrict__ b) {
  int lane  = threadIdx.x & 31;
  int laneN = lane & 15;
  float gc[8], bc[8];
#pragma unroll
  for (int t = 0; t < 8; ++t) { gc[t] = g[t * 16 + laneN]; bc[t] = b[t * 16 + laneN]; }
#pragma unroll
  for (int r = 0; r < 8; ++r) {
    float s1 = 0.f, s2 = 0.f;
#pragma unroll
    for (int t = 0; t < 8; ++t) { float v = acc[t][r]; s1 += v; s2 += v * v; }
#pragma unroll
    for (int off = 1; off < 16; off <<= 1) {   // row lives in one 16-lane half
      s1 += __shfl_xor(s1, off, 32);
      s2 += __shfl_xor(s2, off, 32);
    }
    float mean = s1 * (1.0f / 128.0f);
    float var  = s2 * (1.0f / 128.0f) - mean * mean;
    float rstd = rsqrtf(var + 1e-5f);
    int m = (lane < 16) ? r : (r + 8);
    float* orow = outBase + (long long)(rowBase + m) * NF;
#pragma unroll
    for (int t = 0; t < 8; ++t)
      orow[t * 16 + laneN] = (acc[t][r] - mean) * rstd * gc[t] + bc[t];
  }
}

// ---------------------------------------------------------------------------
// Stage 0: Cb virtual atom
// ---------------------------------------------------------------------------
__global__ void k_cb(const float* __restrict__ X, float* __restrict__ Cb) {
  int i = blockIdx.x * blockDim.x + threadIdx.x;
  if (i >= LRES) return;
  const float* xr = X + (long long)i * 15;
  float nx = xr[0], ny = xr[1], nz = xr[2];
  float ax = xr[3], ay = xr[4], az = xr[5];      // Ca
  float cx = xr[6], cy = xr[7], cz = xr[8];      // C
  float bx = ax - nx, by = ay - ny, bz = az - nz;
  float gx = cx - ax, gy = cy - ay, gz = cz - az;
  float px = by * gz - bz * gy;
  float py = bz * gx - bx * gz;
  float pz = bx * gy - by * gx;
  Cb[i * 3 + 0] = -0.58273431f * px + 0.56802827f * bx - 0.54067466f * gx + ax;
  Cb[i * 3 + 1] = -0.58273431f * py + 0.56802827f * by - 0.54067466f * gy + ay;
  Cb[i * 3 + 2] = -0.58273431f * pz + 0.56802827f * bz - 0.54067466f * gz + az;
}

// ---------------------------------------------------------------------------
// Stage 0b: f32 -> bf16 weight conversion (+ K padding)
// ---------------------------------------------------------------------------
__global__ void k_convert_w(const float* __restrict__ We, const float* __restrict__ Wn,
                            const float* __restrict__ Wj,
                            bf16_t* __restrict__ WbE, bf16_t* __restrict__ WbN,
                            bf16_t* __restrict__ WbJ) {
  int tid = blockIdx.x * blockDim.x + threadIdx.x;
  if (tid < KE * NF) WbE[tid] = (bf16_t)We[tid];
  if (tid < KV * NF) {
    int k = tid >> 7;
    WbN[tid] = (k < 185) ? (bf16_t)Wn[tid] : (bf16_t)0.0f;
  }
  if (tid < KJ * NF) {
    int k = tid >> 7;
    WbJ[tid] = (k < 16) ? (bf16_t)Wj[tid] : (bf16_t)0.0f;
  }
}

// ---------------------------------------------------------------------------
// Stage 1: top-30 Ca neighbors per residue (block per residue, LDS argmin)
// ---------------------------------------------------------------------------
__global__ void k_topk_nb(const float* __restrict__ X, float* __restrict__ Dnb,
                          int* __restrict__ EidxOut) {
  __shared__ float d[LRES];
  __shared__ float rv[256];
  __shared__ int   ri[256];
  int i = blockIdx.x, tid = threadIdx.x;
  const float* cai = X + (long long)(i * 5 + 1) * 3;
  float cx = cai[0], cy = cai[1], cz = cai[2];
  for (int j = tid; j < LRES; j += 256) {
    const float* cj = X + (long long)(j * 5 + 1) * 3;
    float dx = cx - cj[0], dy = cy - cj[1], dz = cz - cj[2];
    d[j] = sqrtf(dx * dx + dy * dy + dz * dz + 1e-6f);
  }
  __syncthreads();
  for (int sel = 0; sel < KNB; ++sel) {
    float bv = 3.0e38f; int bi = 0x7fffffff;
    for (int j = tid; j < LRES; j += 256) {
      float v = d[j];
      if (v < bv || (v == bv && j < bi)) { bv = v; bi = j; }
    }
    rv[tid] = bv; ri[tid] = bi;
    __syncthreads();
    for (int s = 128; s > 0; s >>= 1) {
      if (tid < s) {
        float ov = rv[tid + s]; int oi = ri[tid + s];
        if (ov < rv[tid] || (ov == rv[tid] && oi < ri[tid])) { rv[tid] = ov; ri[tid] = oi; }
      }
      __syncthreads();
    }
    if (tid == 0) {
      int w = ri[0];
      EidxOut[i * KNB + sel] = w;
      Dnb[i * KNB + sel] = rv[0];
      d[w] = 3.0e38f;
    }
    __syncthreads();
  }
}

// ---------------------------------------------------------------------------
// Stage 2: edge feature rows (416 bf16 per (i,k)); wave per row
// ---------------------------------------------------------------------------
__global__ void k_edge_feats(const float* __restrict__ X, const float* __restrict__ Cb,
                             const float* __restrict__ Dnb, const int* __restrict__ Eidx,
                             const int* __restrict__ resi,
                             const float* __restrict__ Wpos, const float* __restrict__ bpos,
                             bf16_t* __restrict__ feats) {
  __shared__ float pd[8][26];
  int tid = threadIdx.x;
  int wave = tid >> 5, lane = tid & 31;
  int row = blockIdx.x * 8 + wave;               // 30720 rows
  int i = row / KNB;
  int j = Eidx[row];
  if (lane < 25) {
    float dv;
    if (lane == 0) {
      dv = Dnb[row];
    } else {
      float3 a = get_atom(X, Cb, i, c_pa[lane]);
      float3 b = get_atom(X, Cb, j, c_pb[lane]);
      float dx = a.x - b.x, dy = a.y - b.y, dz = a.z - b.z;
      dv = sqrtf(dx * dx + dy * dy + dz * dz + 1e-6f);
    }
    pd[wave][lane] = dv;
  }
  __syncthreads();
  int dlt = resi[i] - resi[j] + 32;
  dlt = dlt < 0 ? 0 : (dlt > 64 ? 64 : dlt);
  bf16_t* frow = feats + (long long)row * KE;
  for (int e = lane; e < KE; e += 32) {
    float v;
    if (e < 16) {
      v = Wpos[dlt * 16 + e] + bpos[e];
    } else {
      int p = (e - 16) >> 4;
      int m = (e - 16) & 15;
      v = rbf_val(pd[wave][p], m);
    }
    frow[e] = (bf16_t)v;
  }
}

// ---------------------------------------------------------------------------
// Stage 3: ligand-context top-25 per residue
// ---------------------------------------------------------------------------
__global__ void k_ctx_topk(const float* __restrict__ Cb,
                           const float* __restrict__ Y, const float* __restrict__ Ym,
                           const float* __restrict__ Z, const float* __restrict__ Zm,
                           const int* __restrict__ Zt, const float* __restrict__ mask,
                           float* __restrict__ Jp, int* __restrict__ Jt,
                           float* __restrict__ Meff) {
  __shared__ float dj[272], px[272], py[272], pz[272], mj[272];
  __shared__ int   tj[272];
  __shared__ float rv[256];
  __shared__ int   ri[256];
  int i = blockIdx.x, tid = threadIdx.x;
  float cbx = Cb[i * 3], cby = Cb[i * 3 + 1], cbz = Cb[i * 3 + 2];
  float mi = mask[i];
  for (int m = tid; m < 272; m += 256) {
    float x, y, z, mv; int t;
    if (m < 208) {
      const float* p = Y + ((long long)i * 208 + m) * 3;
      x = p[0]; y = p[1]; z = p[2];
      mv = Ym[(long long)i * 208 + m];
      t = c_dnarna[m % 26];
    } else {
      int mz = m - 208;
      x = Z[mz * 3]; y = Z[mz * 3 + 1]; z = Z[mz * 3 + 2];
      mv = Zm[mz]; t = Zt[mz];
    }
    float mJ = mi * mv;
    float dx = cbx - x, dy = cby - y, dz = cbz - z;
    float D = sqrtf(dx * dx + dy * dy + dz * dz + 1e-6f);
    dj[m] = D * mJ + (1.0f - mJ) * 1.0e4f;
    px[m] = x; py[m] = y; pz[m] = z; tj[m] = t; mj[m] = mJ;
  }
  __syncthreads();
  for (int sel = 0; sel < CTX; ++sel) {
    float bv = 3.0e38f; int bi = 0x7fffffff;
    for (int m = tid; m < 272; m += 256) {
      float v = dj[m];
      if (v < bv || (v == bv && m < bi)) { bv = v; bi = m; }
    }
    rv[tid] = bv; ri[tid] = bi;
    __syncthreads();
    for (int s = 128; s > 0; s >>= 1) {
      if (tid < s) {
        float ov = rv[tid + s]; int oi = ri[tid + s];
        if (ov < rv[tid] || (ov == rv[tid] && oi < ri[tid])) { rv[tid] = ov; ri[tid] = oi; }
      }
      __syncthreads();
    }
    if (tid == 0) {
      int w = ri[0];
      int o = i * CTX + sel;
      Jp[o * 3]     = px[w];
      Jp[o * 3 + 1] = py[w];
      Jp[o * 3 + 2] = pz[w];
      Jt[o] = tj[w];
      Meff[o] = mj[w] * (rv[0] < 20.0f ? 1.0f : 0.0f);
      dj[w] = 3.0e38f;
    }
    __syncthreads();
  }
}

// ---------------------------------------------------------------------------
// Stage 4: node feature rows (192 bf16 per (i,c)); wave per row
// ---------------------------------------------------------------------------
__global__ void k_node_feats(const float* __restrict__ X, const float* __restrict__ Cb,
                             const float* __restrict__ Jp, const int* __restrict__ Jt,
                             const float* __restrict__ Meff, bf16_t* __restrict__ feats) {
  int tid = threadIdx.x;
  int wave = tid >> 5, lane = tid & 31;
  int row = blockIdx.x * 8 + wave;               // 25600 rows
  int i = row / CTX;
  const float* jp = Jp + (long long)row * 3;
  float jx = jp[0], jy = jp[1], jz = jp[2];
  float dist[5];
#pragma unroll
  for (int a = 0; a < 5; ++a) {
    float3 p = get_atom(X, Cb, i, a);
    float dx = p.x - jx, dy = p.y - jy, dz = p.z - jz;
    dist[a] = sqrtf(dx * dx + dy * dy + dz * dz + 1e-6f);
  }
  float s = Meff[row];
  int t = Jt[row];
  bf16_t* frow = feats + (long long)row * KV;
  for (int e = lane; e < KV; e += 32) {
    float v = 0.0f;
    if (e < 80) {
      int a = e >> 4, m = e & 15;
      float da = (a == 0) ? dist[0] : (a == 1) ? dist[1] : (a == 2) ? dist[2]
                 : (a == 3) ? dist[3] : dist[4];
      v = rbf_val(da, m) * s;
    } else if (e < 185) {
      v = (e - 80 == t) ? s : 0.0f;
    }
    frow[e] = (bf16_t)v;
  }
}

// ---------------------------------------------------------------------------
// Stage 5: bf16 WMMA GEMM (MxKPAD @ KPADx128) with fused LayerNorm epilogue.
// Whole B staged once (transposed) in big CDNA5 LDS; per k-step: 2x
// global_load_b128 (next A, software-pipelined), 16x ds_load_b128 into
// distinct regs, then 8 back-to-back WMMAs on independent accumulators.
// ---------------------------------------------------------------------------
template <int KPAD>
__global__ void k_gemm_ln(const bf16_t* __restrict__ A, const bf16_t* __restrict__ Bw,
                          const float* __restrict__ g, const float* __restrict__ b,
                          float* __restrict__ out) {
  __shared__ bf16_t sB[KPAD * NF];               // [n][k], 2*KPAD*128 bytes (<=104KB)
  int tid = threadIdx.x;
  for (int e = tid; e < KPAD * NF; e += 256) {
    int n = e & 127, k = e >> 7;                 // coalesced global read over n
    sB[n * KPAD + k] = Bw[k * NF + n];
  }
  __syncthreads();

  int wave = tid >> 5, lane = tid & 31;
  int laneN = lane & 15;
  int tileM = blockIdx.x * 8 + wave;
  int rowBase = tileM * 16;
  int khalf = (lane >> 4) ? 8 : 0;               // A: lane halves carry K 0..7 / 8..15
  int bhalf = (lane >> 4) ? 8 : 0;               // B: lane halves carry K 0..7,16..23 / 8..15,24..31
  const bf16_t* aRow = A + (long long)(rowBase + laneN) * KPAD;

  v8f acc[8] = {};
  AF af;                                         // software-pipelined A fragment
  af.h[0] = *(const v8bf*)(aRow + khalf);
  af.h[1] = *(const v8bf*)(aRow + 16 + khalf);

#pragma unroll
  for (int k0 = 0; k0 < KPAD; k0 += 32) {
    AF cur = af;
    if (k0 + 32 < KPAD) {                        // prefetch next A fragment
      af.h[0] = *(const v8bf*)(aRow + k0 + 32 + khalf);
      af.h[1] = *(const v8bf*)(aRow + k0 + 48 + khalf);
    }
    AF bfr[8];                                   // all 8 B fragments up front
#pragma unroll
    for (int t = 0; t < 8; ++t) {
      const bf16_t* bp = &sB[(t * 16 + laneN) * KPAD + k0 + bhalf];
      bfr[t].h[0] = *(const v8bf*)(bp);
      bfr[t].h[1] = *(const v8bf*)(bp + 16);
    }
#pragma unroll
    for (int t = 0; t < 8; ++t)
      acc[t] = __builtin_amdgcn_wmma_f32_16x16x32_bf16(
          false, cur.v, false, bfr[t].v, (short)0, acc[t], false, false);
  }
  ln_store_strip(out, rowBase, acc, g, b);
}

// ---------------------------------------------------------------------------
// Stage 6: J_edges — RBF features built IN REGISTERS (K=16 pad 32), one WMMA
// per 16x16 tile, fused LayerNorm, direct store (single pass over 328 MB).
// B fragment loads issued first; exp() VALU work hides the LDS latency.
// ---------------------------------------------------------------------------
__global__ void k_jedges(const float* __restrict__ Jp, const bf16_t* __restrict__ Wbj,
                         const float* __restrict__ g, const float* __restrict__ b,
                         float* __restrict__ out) {
  __shared__ bf16_t sB[NF * KJ];
  int tid = threadIdx.x;
  for (int e = tid; e < NF * KJ; e += 256) {
    int n = e & 127, k = e >> 7;
    sB[n * KJ + k] = Wbj[k * NF + n];
  }
  __syncthreads();
  int wave = tid >> 5, lane = tid & 31, laneN = lane & 15;
  int bhalf = (lane >> 4) ? 8 : 0;
  int tileM = blockIdx.x * 8 + wave;

  AF bfr[8];                                     // issue LDS loads first
#pragma unroll
  for (int t = 0; t < 8; ++t) {
    const bf16_t* bp = &sB[(t * 16 + laneN) * KJ + bhalf];
    bfr[t].h[0] = *(const v8bf*)(bp);
    bfr[t].h[1] = *(const v8bf*)(bp + 16);
  }

  int row = tileM * 16 + laneN;                  // both lane halves: same M row
  int i = row / 625, rem = row % 625;
  int c1 = rem / 25, c2 = rem % 25;
  const float* p1 = Jp + (long long)(i * 25 + c1) * 3;
  const float* p2 = Jp + (long long)(i * 25 + c2) * 3;
  float dx = p1[0] - p2[0], dy = p1[1] - p2[1], dz = p1[2] - p2[2];
  float d = sqrtf(dx * dx + dy * dy + dz * dz + 1e-6f);
  int mu0 = (lane >> 4) * 8;                     // lanes 0-15: K 0..7; 16-31: K 8..15
  AF af;
#pragma unroll
  for (int q = 0; q < 8; ++q) {
    af.e[q]     = (bf16_t)rbf_val(d, mu0 + q);
    af.e[q + 8] = (bf16_t)0.0f;                  // K padding 16..31
  }

  v8f acc[8] = {};
#pragma unroll
  for (int t = 0; t < 8; ++t)
    acc[t] = __builtin_amdgcn_wmma_f32_16x16x32_bf16(
        false, af.v, false, bfr[t].v, (short)0, acc[t], false, false);
  ln_store_strip(out, tileM * 16, acc, g, b);
}

// ---------------------------------------------------------------------------
// Stage 7: J_nodes — one-hot @ W_jn == row gather, fused LayerNorm
// ---------------------------------------------------------------------------
__global__ void k_jnodes(const int* __restrict__ Jt, const float* __restrict__ Wjn,
                         const float* __restrict__ g, const float* __restrict__ b,
                         float* __restrict__ out) {
  int tid = threadIdx.x;
  int wave = tid >> 5, lane = tid & 31;
  int row = blockIdx.x * 8 + wave;               // 25600 rows
  int t = Jt[row];
  const float* wr = Wjn + (long long)t * NF;
  float x[4];
#pragma unroll
  for (int q = 0; q < 4; ++q) x[q] = wr[lane + 32 * q];
  float s1 = 0.f, s2 = 0.f;
#pragma unroll
  for (int q = 0; q < 4; ++q) { s1 += x[q]; s2 += x[q] * x[q]; }
#pragma unroll
  for (int off = 1; off < 32; off <<= 1) {
    s1 += __shfl_xor(s1, off, 32);
    s2 += __shfl_xor(s2, off, 32);
  }
  float mean = s1 * (1.0f / 128.0f);
  float var  = s2 * (1.0f / 128.0f) - mean * mean;
  float rstd = rsqrtf(var + 1e-5f);
  float* orow = out + (long long)row * NF;
#pragma unroll
  for (int q = 0; q < 4; ++q) {
    int c = lane + 32 * q;
    orow[c] = (x[q] - mean) * rstd * g[c] + b[c];
  }
}

// ---------------------------------------------------------------------------
// Launch
// ---------------------------------------------------------------------------
extern "C" void kernel_launch(void* const* d_in, const int* in_sizes, int n_in,
                              void* d_out, int out_size, void* d_ws, size_t ws_size,
                              hipStream_t stream) {
  const float* Z      = (const float*)d_in[0];
  const float* Z_m    = (const float*)d_in[1];
  const int*   Z_t    = (const int*)  d_in[2];
  const float* X      = (const float*)d_in[3];
  const float* Y      = (const float*)d_in[4];
  const float* Y_m    = (const float*)d_in[5];
  const float* mask   = (const float*)d_in[6];
  const int*   resi   = (const int*)  d_in[8];
  const float* W_pos  = (const float*)d_in[10];
  const float* b_pos  = (const float*)d_in[11];
  const float* W_edge = (const float*)d_in[12];
  const float* g_edge = (const float*)d_in[13];
  const float* b_edge = (const float*)d_in[14];
  const float* W_node = (const float*)d_in[15];
  const float* g_node = (const float*)d_in[16];
  const float* b_node = (const float*)d_in[17];
  const float* W_jn   = (const float*)d_in[18];
  const float* g_jn   = (const float*)d_in[19];
  const float* b_jn   = (const float*)d_in[20];
  const float* W_je   = (const float*)d_in[21];
  const float* g_je   = (const float*)d_in[22];
  const float* b_je   = (const float*)d_in[23];

  float* outF = (float*)d_out;
  int*   Eidx = (int*)(outF + OFF_EIDX);

  // workspace carve-out
  char* w = (char*)d_ws;
  auto alloc = [&](size_t bytes) -> void* {
    void* p = (void*)w;
    w += (bytes + 255) & ~(size_t)255;
    return p;
  };
  float*  Cb     = (float*) alloc((size_t)LRES * 3 * 4);
  float*  Dnb    = (float*) alloc((size_t)ROWS_E * 4);
  float*  Jp     = (float*) alloc((size_t)ROWS_V * 3 * 4);
  int*    Jt     = (int*)   alloc((size_t)ROWS_V * 4);
  float*  Meff   = (float*) alloc((size_t)ROWS_V * 4);
  bf16_t* featsE = (bf16_t*)alloc((size_t)ROWS_E * KE * 2);
  bf16_t* featsV = (bf16_t*)alloc((size_t)ROWS_V * KV * 2);
  bf16_t* WbE    = (bf16_t*)alloc((size_t)KE * NF * 2);
  bf16_t* WbN    = (bf16_t*)alloc((size_t)KV * NF * 2);
  bf16_t* WbJ    = (bf16_t*)alloc((size_t)KJ * NF * 2);

  k_cb<<<(LRES + 255) / 256, 256, 0, stream>>>(X, Cb);
  k_convert_w<<<(KE * NF + 255) / 256, 256, 0, stream>>>(W_edge, W_node, W_je,
                                                         WbE, WbN, WbJ);
  k_topk_nb<<<LRES, 256, 0, stream>>>(X, Dnb, Eidx);
  k_edge_feats<<<ROWS_E / 8, 256, 0, stream>>>(X, Cb, Dnb, Eidx, resi,
                                               W_pos, b_pos, featsE);
  k_gemm_ln<KE><<<ROWS_E / 128, 256, 0, stream>>>(featsE, WbE, g_edge, b_edge,
                                                  outF + OFF_E);
  k_ctx_topk<<<LRES, 256, 0, stream>>>(Cb, Y, Y_m, Z, Z_m, Z_t, mask,
                                       Jp, Jt, Meff);
  k_node_feats<<<ROWS_V / 8, 256, 0, stream>>>(X, Cb, Jp, Jt, Meff, featsV);
  k_gemm_ln<KV><<<ROWS_V / 128, 256, 0, stream>>>(featsV, WbN, g_node, b_node,
                                                  outF + OFF_V);
  k_jnodes<<<ROWS_V / 8, 256, 0, stream>>>(Jt, W_jn, g_jn, b_jn, outF + OFF_JN);
  k_jedges<<<ROWS_JE / 128, 256, 0, stream>>>(Jp, WbJ, g_je, b_je, outF + OFF_JE);
}
